// S6Block_69295002354167
// MI455X (gfx1250) — matrix operand order repaired and checked by reading, not compiled, not run on hardware
//
#include <hip/hip_runtime.h>

// S6 (Mamba) block for MI455X / gfx1250, wave32, WMMA f32_16x16x32_f16.
//
// Pipeline (all on `stream`):
//  1) convert x -> f16; transpose+convert W_in, W_x, W_out -> f16 (B^T layout)
//  2) A_discrete = exp(-exp(A_log))
//  3) GEMM1 (WMMA): [8192,1024]f16 @ [1024,2048] -> split f32 x_ssm / gate
//  4) depthwise causal conv (K=4) + SiLU -> x_conv f32 + f16 copy
//  5) GEMM2 (WMMA): x_conv_h @ W_x -> BC [8192,32] f32
//  6) selective scan: 2048 channels, N=16 state in VGPRs, B/C staged in LDS
//  7) gate: y * silu(gate) -> f16
//  8) GEMM3 (WMMA): -> y [8192,1024] f32 into d_out
//  9) final_state = mean_d h  (deterministic 2-stage reduction)

typedef _Float16 half_t;
typedef __attribute__((ext_vector_type(8)))  _Float16 v8h;
typedef __attribute__((ext_vector_type(16))) _Float16 v16h;
typedef __attribute__((ext_vector_type(8)))  float    v8f;

// ---------------------------------------------------------------- elementwise
__global__ void cvt_f16_kernel(const float* __restrict__ src,
                               half_t* __restrict__ dst, int n) {
  int i = blockIdx.x * blockDim.x + threadIdx.x;
  if (i < n) dst[i] = (half_t)src[i];
}

// W [K,N] f32 row-major  ->  WT [N,K] f16 row-major (i.e. B^T for WMMA B-frags)
__global__ void transpose_cvt_kernel(const float* __restrict__ W,
                                     half_t* __restrict__ WT, int K, int N) {
  int i = blockIdx.x * blockDim.x + threadIdx.x;
  if (i < K * N) {
    int n = i / K, k = i % K;
    WT[i] = (half_t)W[(size_t)k * N + n];
  }
}

__global__ void adisc_kernel(const float* __restrict__ A_log,
                             float* __restrict__ Ad, int n) {
  int i = blockIdx.x * blockDim.x + threadIdx.x;
  if (i < n) Ad[i] = expf(-expf(A_log[i]));
}

// ------------------------------------------------------------------ WMMA GEMM
// C[M,N] = A[M,K](f16,row-major) * B (given as BT[N,K] f16 row-major) + bias
// One wave computes a 16 x (16*NT) tile; A-fragment reused across NT B-frags.
// SPLIT=1: columns < splitCol -> out0 (stride splitCol), rest -> out1.
template <int NT, int SPLIT>
__global__ __launch_bounds__(256) void gemm_wmma_kernel(
    const half_t* __restrict__ A, const half_t* __restrict__ BT,
    const float* __restrict__ bias, float* __restrict__ out0,
    float* __restrict__ out1, int M, int N, int K, int splitCol) {
  const int lane = threadIdx.x & 31;
  const int wave = blockIdx.x * (blockDim.x >> 5) + (threadIdx.x >> 5);
  const int mTiles = M >> 4;
  const int nGroups = N / (16 * NT);
  if (wave >= mTiles * nGroups) return;
  const int tileM = wave % mTiles;
  const int group = wave / mTiles;
  const int row0 = tileM << 4;

  const int mr = lane & 15;   // row (A) / column (B) owned by this lane
  const int hi = lane >> 4;   // lane-half selects K sub-ranges
  const half_t* Arow = A + (size_t)(row0 + mr) * K;

  v8f acc[NT];
#pragma unroll
  for (int j = 0; j < NT; ++j) {
#pragma unroll
    for (int r = 0; r < 8; ++r) acc[j][r] = 0.0f;
  }

  for (int k0 = 0; k0 < K; k0 += 32) {
    // A 16x32 frag: lane<16 holds K{0..15}, lane>=16 holds K{16..31}
    const v16h afrag = *reinterpret_cast<const v16h*>(Arow + k0 + hi * 16);
#pragma unroll
    for (int j = 0; j < NT; ++j) {
      // B 32x16 frag from BT: lane owns column n=mr; lane-half picks K 0..15 / 16..31
      const int ncol = (group * NT + j) * 16 + mr;
      const v16h bfrag =
          *reinterpret_cast<const v16h*>(BT + (size_t)ncol * K + k0 + hi * 16);
      acc[j] = __builtin_amdgcn_wmma_f32_16x16x32_f16(
          false, afrag, false, bfrag, (short)0, acc[j], false, false);
    }
  }

  // C/D layout: VGPR r, lanes 0-15 -> row r, lanes 16-31 -> row 8+r; col = lane&15
  const int colLane = lane & 15;
  const int rowOff = hi * 8;
#pragma unroll
  for (int j = 0; j < NT; ++j) {
    const int col = (group * NT + j) * 16 + colLane;
    const float bv = bias[col];
    if (SPLIT && col >= splitCol) {
      const int c2 = col - splitCol;
      const int stride = N - splitCol;
#pragma unroll
      for (int r = 0; r < 8; ++r)
        out1[(size_t)(row0 + rowOff + r) * stride + c2] = acc[j][r] + bv;
    } else {
      const int stride = SPLIT ? splitCol : N;
#pragma unroll
      for (int r = 0; r < 8; ++r)
        out0[(size_t)(row0 + rowOff + r) * stride + col] = acc[j][r] + bv;
    }
  }
}

// ------------------------------------------------- depthwise causal conv+SiLU
__global__ void conv_silu_kernel(const float* __restrict__ xs,
                                 const float* __restrict__ cw,
                                 const float* __restrict__ cb,
                                 float* __restrict__ xc,
                                 half_t* __restrict__ xch, int total, int Dd,
                                 int Ls) {
  int i = blockIdx.x * blockDim.x + threadIdx.x;
  if (i >= total) return;
  int d = i % Dd;
  int t = (i / Dd) % Ls;
  float acc = cb[d];
#pragma unroll
  for (int j = 0; j < 4; ++j) {
    int tt = t - 3 + j;
    if (tt >= 0) acc += xs[i + (j - 3) * Dd] * cw[d * 4 + j];
  }
  float s = acc / (1.0f + expf(-acc));  // silu
  xc[i] = s;
  xch[i] = (half_t)s;
}

// ------------------------------------------------------------- selective scan
// One thread per (b,d) channel; state h[16] in VGPRs; B_t/C_t staged in LDS
// in 64-step chunks (uniform b per block since 256 | D).
__global__ __launch_bounds__(256) void scan_kernel(
    const float* __restrict__ xc, const float* __restrict__ BC,
    const float* __restrict__ Ad, const float* __restrict__ Dp,
    float* __restrict__ y, float* __restrict__ hout, int Ls, int Dd) {
  const int tid = threadIdx.x;
  const int gid = blockIdx.x * 256 + tid;
  const int b = gid / Dd;
  const int d = gid % Dd;
  __shared__ float sBC[64 * 32];

  float h[16], Av[16];
#pragma unroll
  for (int n = 0; n < 16; ++n) {
    h[n] = 0.0f;
    Av[n] = Ad[d * 16 + n];
  }
  const float Dv = Dp[d];
  const float* xrow = xc + (size_t)b * Ls * Dd + d;
  float* yrow = y + (size_t)b * Ls * Dd + d;
  const float* bcrow = BC + (size_t)b * Ls * 32;

  for (int tb = 0; tb < Ls; tb += 64) {
    __syncthreads();
    for (int i = tid; i < 64 * 32; i += 256) sBC[i] = bcrow[(size_t)tb * 32 + i];
    __syncthreads();
    for (int tt = 0; tt < 64; ++tt) {
      const int t = tb + tt;
      const float xv = xrow[(size_t)t * Dd];
      const float* bc = &sBC[tt * 32];
      float yv = Dv * xv;
#pragma unroll
      for (int n = 0; n < 16; ++n) {
        h[n] = Av[n] * h[n] + xv * bc[n];
        yv += bc[16 + n] * h[n];
      }
      yrow[(size_t)t * Dd] = yv;
    }
  }
#pragma unroll
  for (int n = 0; n < 16; ++n) hout[(size_t)gid * 16 + n] = h[n];
}

// --------------------------------------------------------------- gate epilog
__global__ void gate_kernel(const float* __restrict__ ys,
                            const float* __restrict__ g,
                            half_t* __restrict__ out, int n) {
  int i = blockIdx.x * blockDim.x + threadIdx.x;
  if (i < n) {
    float gv = g[i];
    out[i] = (half_t)(ys[i] * gv / (1.0f + expf(-gv)));
  }
}

// -------------------------------------------- final_state = mean_d h[b,d,n]
__global__ __launch_bounds__(256) void fstate_kernel(
    const float* __restrict__ hout, float* __restrict__ fs, int Dd) {
  const int blk = blockIdx.x;  // b*16 + n
  const int b = blk >> 4;
  const int n = blk & 15;
  __shared__ float red[256];
  float s = 0.0f;
  for (int d = threadIdx.x; d < Dd; d += 256)
    s += hout[(size_t)(b * Dd + d) * 16 + n];
  red[threadIdx.x] = s;
  __syncthreads();
  for (int off = 128; off > 0; off >>= 1) {
    if (threadIdx.x < off) red[threadIdx.x] += red[threadIdx.x + off];
    __syncthreads();
  }
  if (threadIdx.x == 0) fs[blk] = red[0] / (float)Dd;
}

// ---------------------------------------------------------------------- host
extern "C" void kernel_launch(void* const* d_in, const int* in_sizes, int n_in,
                              void* d_out, int out_size, void* d_ws,
                              size_t ws_size, hipStream_t stream) {
  constexpr int BB = 2, LL = 4096, DD = 1024;
  constexpr int BL = BB * LL;  // 8192
  constexpr size_t MB = 1024ull * 1024ull;

  const float* x      = (const float*)d_in[0];
  const float* W_in   = (const float*)d_in[1];
  const float* b_in   = (const float*)d_in[2];
  const float* conv_w = (const float*)d_in[3];
  const float* conv_b = (const float*)d_in[4];
  const float* W_x    = (const float*)d_in[5];
  const float* b_x    = (const float*)d_in[6];
  const float* A_log  = (const float*)d_in[7];
  const float* D_par  = (const float*)d_in[8];
  const float* W_out  = (const float*)d_in[9];
  const float* b_out  = (const float*)d_in[10];

  // Workspace layout (~138 MB peak, with reuse):
  char* ws = (char*)d_ws;
  half_t* XH     = (half_t*)(ws + 0);        // 16MB  x f16 (reused as YGH)
  half_t* WIN_T  = (half_t*)(ws + 16 * MB);  // 4MB   W_in^T f16
  half_t* WOUT_T = (half_t*)(ws + 20 * MB);  // 2MB   W_out^T f16
  half_t* WX_T   = (half_t*)(ws + 22 * MB);  // 64KB  W_x^T f16
  float*  ADISC  = (float*)(ws + 23 * MB);   // 64KB  exp(-exp(A_log))
  float*  XSSM   = (float*)(ws + 24 * MB);   // 32MB  x_ssm f32 (reused as YSCAN)
  float*  GATE   = (float*)(ws + 56 * MB);   // 32MB  gate f32
  float*  XCONV  = (float*)(ws + 88 * MB);   // 32MB  x_conv f32
  half_t* XCONVH = (half_t*)(ws + 120 * MB); // 16MB  x_conv f16
  float*  BCbuf  = (float*)(ws + 136 * MB);  // 1MB   BC f32
  float*  HOUT   = (float*)(ws + 137 * MB);  // 128KB h_final per channel
  half_t* YGH    = XH;                       // reuse: gated y, f16
  float*  YSCAN  = XSSM;                     // reuse: scan output, f32

  float* yout = (float*)d_out;               // [B,L,D] f32
  float* fsout = yout + (size_t)BB * LL * DD;  // [B,N] f32

  const int nElem = BL * DD;  // 8,388,608

  // 1) precision converts + A_discrete
  cvt_f16_kernel<<<(nElem + 255) / 256, 256, 0, stream>>>(x, XH, nElem);
  transpose_cvt_kernel<<<(DD * 2 * DD + 255) / 256, 256, 0, stream>>>(
      W_in, WIN_T, DD, 2 * DD);
  transpose_cvt_kernel<<<(DD * 32 + 255) / 256, 256, 0, stream>>>(W_x, WX_T,
                                                                  DD, 32);
  transpose_cvt_kernel<<<(DD * DD + 255) / 256, 256, 0, stream>>>(W_out,
                                                                  WOUT_T, DD, DD);
  adisc_kernel<<<(DD * 16 + 255) / 256, 256, 0, stream>>>(A_log, ADISC, DD * 16);

  // 2) GEMM1: x @ W_in + b_in  -> split x_ssm / gate
  {
    const int waves = (BL / 16) * ((2 * DD) / 64);  // 16384
    gemm_wmma_kernel<4, 1><<<waves / 8, 256, 0, stream>>>(
        XH, WIN_T, b_in, XSSM, GATE, BL, 2 * DD, DD, DD);
  }

  // 3) depthwise causal conv + SiLU
  conv_silu_kernel<<<(nElem + 255) / 256, 256, 0, stream>>>(
      XSSM, conv_w, conv_b, XCONV, XCONVH, nElem, DD, LL);

  // 4) GEMM2: x_conv @ W_x + b_x -> BC
  {
    const int waves = (BL / 16) * (32 / 32);  // 512
    gemm_wmma_kernel<2, 0><<<waves / 8, 256, 0, stream>>>(
        XCONVH, WX_T, b_x, BCbuf, nullptr, BL, 32, DD, 0);
  }

  // 5) selective scan (overwrites XSSM region with y_scan)
  scan_kernel<<<(BB * DD) / 256, 256, 0, stream>>>(XCONV, BCbuf, ADISC, D_par,
                                                   YSCAN, HOUT, LL, DD);

  // 6) gating -> f16 (reuses XH region)
  gate_kernel<<<(nElem + 255) / 256, 256, 0, stream>>>(YSCAN, GATE, YGH, nElem);

  // 7) GEMM3: y_gated @ W_out + b_out -> d_out
  {
    const int waves = (BL / 16) * (DD / 64);  // 8192
    gemm_wmma_kernel<4, 0><<<waves / 8, 256, 0, stream>>>(
        YGH, WOUT_T, b_out, yout, nullptr, BL, DD, DD, 0);
  }

  // 8) final_state = mean over D of h
  fstate_kernel<<<BB * 16, 256, 0, stream>>>(HOUT, fsout, DD);
}
